// SpaceTimeTransformerBlock_56066503082212
// MI455X (gfx1250) — matrix operand order repaired and checked by reading, not compiled, not run on hardware
//
#include <hip/hip_runtime.h>
#include <hip/hip_bf16.h>
#include <stdint.h>
#include <math.h>

typedef __attribute__((ext_vector_type(16))) _Float16 v16h;
typedef __attribute__((ext_vector_type(8)))  float    v8f;

#define D_    1024
#define NH_   16
#define HD_   64
#define GRID2 24
#define N_    (GRID2 * GRID2)   /* 576 */
#define BT_   16
#define MTOK_ (BT_ * N_)        /* 9216 */
#define HID_  4096
#define EPS_  1e-6f
#define CAP_  50.0f

// ---------------------------------------------------------------- async LDS path
// GLOBAL_LOAD_ASYNC_TO_LDS_B128 (ISA 15.18.3 op 98): VDST = LDS byte-address VGPR,
// VADDR = 64-bit global address, saddr = off. Tracked by ASYNCcnt.
#define ASYNC_LDS 1

__device__ __forceinline__ void async_copy16(void* lds_dst, const void* gsrc) {
  uint32_t lds_off = (uint32_t)(uintptr_t)lds_dst;   // generic LDS addr low 32 bits
  uint64_t gaddr = (uint64_t)(uintptr_t)gsrc;
  asm volatile("global_load_async_to_lds_b128 %0, %1, off"
               :: "v"(lds_off), "v"(gaddr)
               : "memory");
}

__device__ __forceinline__ void wait_async() {
#if __has_builtin(__builtin_amdgcn_s_wait_asynccnt)
  __builtin_amdgcn_s_wait_asynccnt(0);
#else
  asm volatile("s_wait_asynccnt 0x0" ::: "memory");
#endif
}

// ---------------------------------------------------------------- WMMA helpers
__device__ __forceinline__ v8f wmma_f16(v16h a, v16h b, v8f c) {
  return __builtin_amdgcn_wmma_f32_16x16x32_f16(
      /*neg_a=*/false, a, /*neg_b=*/false, b,
      /*c_mod=*/(short)0, c, /*reuse_a=*/false, /*reuse_b=*/false);
}

// A fragment (16x32 f16): lane L -> row M = L%16; element j:
//   k = kbase + (L>=16?16:0) + j
__device__ __forceinline__ v16h frag_a_lds(const _Float16* base, int ldh,
                                           int row, int kbase, int lg) {
  union { uint32_t u[8]; v16h v; } U;
  const uint32_t* p0 = (const uint32_t*)(base + (size_t)row * ldh + kbase + (lg ? 16 : 0));
#pragma unroll
  for (int i = 0; i < 8; ++i) U.u[i] = p0[i];
  return U.v;
}

// B fragment (32x16 f16): lane L -> col N = L%16; element j: k = (L>=16?16:0)+j.
// rowPtr points at 32 contiguous k-values for this N (transposed LDS tile).
__device__ __forceinline__ v16h frag_b_lds(const _Float16* rowPtr, int lg) {
  union { uint32_t u[8]; v16h v; } U;
  const uint32_t* p = (const uint32_t*)(rowPtr + (lg ? 16 : 0));
#pragma unroll
  for (int i = 0; i < 8; ++i) U.u[i] = p[i];
  return U.v;
}

// ---------------------------------------------------------------- fp32 W[K][N] -> f16 Wt[N][K]
__global__ void __launch_bounds__(256)
cvt_transpose_kernel(const float* __restrict__ W, _Float16* __restrict__ Wt,
                     int K, int N) {
  __shared__ _Float16 Ts[32][33];
  int k0 = blockIdx.x * 32, n0 = blockIdx.y * 32;
  int c = threadIdx.x & 31, r = threadIdx.x >> 5;
#pragma unroll
  for (int p = 0; p < 4; ++p) {
    int kr = r + p * 8;
    Ts[kr][c] = (_Float16)W[(size_t)(k0 + kr) * N + n0 + c];
  }
  __syncthreads();
#pragma unroll
  for (int p = 0; p < 4; ++p) {
    int nr = r + p * 8;
    Wt[(size_t)(n0 + nr) * K + k0 + c] = Ts[c][nr];
  }
}

// ---------------------------------------------------------------- RMSNorm -> f16
__global__ void __launch_bounds__(256)
rmsnorm_f16_kernel(const float* __restrict__ x, const float* __restrict__ g,
                   _Float16* __restrict__ out) {
  __shared__ float red[8];
  __shared__ float rs_sh;
  int t = blockIdx.x;
  const float* xr = x + (size_t)t * D_;
  float v[4];
  float ss = 0.f;
#pragma unroll
  for (int i = 0; i < 4; ++i) {
    v[i] = xr[threadIdx.x + i * 256];
    ss += v[i] * v[i];
  }
#pragma unroll
  for (int off = 16; off; off >>= 1) ss += __shfl_xor(ss, off, 32);
  if ((threadIdx.x & 31) == 0) red[threadIdx.x >> 5] = ss;
  __syncthreads();
  float tot = (threadIdx.x < 8) ? red[threadIdx.x] : 0.f;
#pragma unroll
  for (int off = 4; off; off >>= 1) tot += __shfl_xor(tot, off, 32);
  if (threadIdx.x == 0) rs_sh = rsqrtf(tot / (float)D_ + EPS_);
  __syncthreads();
  float rs = rs_sh;
#pragma unroll
  for (int i = 0; i < 4; ++i) {
    int c = threadIdx.x + i * 256;
    out[(size_t)t * D_ + c] = (_Float16)(v[i] * rs * g[c]);
  }
}

// ---------------------------------------------------------------- WMMA GEMM
// C[M,N] = A16[M,K] @ Bt16[N,K]^T (+ resid fp32). Block: 8 waves -> 128x64 tile,
// each wave 16(M) x 64(N): 4 accumulators, 8 WMMAs per 64-wide K step.
template <bool OUT_F16>
__global__ void __launch_bounds__(256)
gemm_wmma_kernel(const _Float16* __restrict__ A, const _Float16* __restrict__ Bt,
                 const float* __restrict__ resid, void* __restrict__ out,
                 int M, int N, int K) {
  __shared__ _Float16 As[128][64];
  __shared__ _Float16 Bs[64][72];
  const int tid = threadIdx.x, lane = tid & 31, w = tid >> 5;
  const int lg = lane >> 4, ln = lane & 15;
  const int m0 = blockIdx.y * 128, n0 = blockIdx.x * 64;
  v8f acc[4] = {};

  for (int kk = 0; kk < K; kk += 64) {
#pragma unroll
    for (int i = 0; i < 4; ++i) {  // A: 1024 x 16B
      int idx = tid + i * 256, row = idx >> 3, c4 = idx & 7;
      const _Float16* src = A + (size_t)(m0 + row) * K + kk + c4 * 8;
#if ASYNC_LDS
      async_copy16(&As[row][c4 * 8], src);
#else
      *(uint4*)&As[row][c4 * 8] = *(const uint4*)src;
#endif
    }
#pragma unroll
    for (int i = 0; i < 2; ++i) {  // B: 512 x 16B
      int idx = tid + i * 256, row = idx >> 3, c4 = idx & 7;
      const _Float16* src = Bt + (size_t)(n0 + row) * K + kk + c4 * 8;
#if ASYNC_LDS
      async_copy16(&Bs[row][c4 * 8], src);
#else
      *(uint4*)&Bs[row][c4 * 8] = *(const uint4*)src;
#endif
    }
    wait_async();
    __syncthreads();

#pragma unroll
    for (int kc = 0; kc < 2; ++kc) {
      v16h af = frag_a_lds(&As[0][0], 64, w * 16 + ln, kc * 32, lg);
#pragma unroll
      for (int s = 0; s < 4; ++s) {
        v16h bf = frag_b_lds(&Bs[s * 16 + ln][kc * 32], lg);
        acc[s] = wmma_f16(af, bf, acc[s]);
      }
    }
    __syncthreads();
  }

#pragma unroll
  for (int s = 0; s < 4; ++s) {
    int n = n0 + s * 16 + ln;
#pragma unroll
    for (int r = 0; r < 8; ++r) {
      int m = m0 + w * 16 + r + 8 * lg;
      size_t idx = (size_t)m * N + n;
      float val = acc[s][r];
      if (resid) val += resid[idx];
      if (OUT_F16) ((_Float16*)out)[idx] = (_Float16)val;
      else         ((float*)out)[idx] = val;
    }
  }
}

// ---------------------------------------------------------------- fused FFN: silu(A@W1) * (A@W3)
__global__ void __launch_bounds__(256)
ffn_gate_wmma_kernel(const _Float16* __restrict__ A, const _Float16* __restrict__ W1t,
                     const _Float16* __restrict__ W3t, _Float16* __restrict__ out,
                     int M, int N, int K) {
  __shared__ _Float16 As[128][64];
  __shared__ _Float16 B1s[64][72];
  __shared__ _Float16 B3s[64][72];
  const int tid = threadIdx.x, lane = tid & 31, w = tid >> 5;
  const int lg = lane >> 4, ln = lane & 15;
  const int m0 = blockIdx.y * 128, n0 = blockIdx.x * 64;
  v8f accU[4] = {}, accT[4] = {};

  for (int kk = 0; kk < K; kk += 64) {
#pragma unroll
    for (int i = 0; i < 4; ++i) {
      int idx = tid + i * 256, row = idx >> 3, c4 = idx & 7;
      const _Float16* src = A + (size_t)(m0 + row) * K + kk + c4 * 8;
#if ASYNC_LDS
      async_copy16(&As[row][c4 * 8], src);
#else
      *(uint4*)&As[row][c4 * 8] = *(const uint4*)src;
#endif
    }
#pragma unroll
    for (int i = 0; i < 2; ++i) {
      int idx = tid + i * 256, row = idx >> 3, c4 = idx & 7;
      const _Float16* s1 = W1t + (size_t)(n0 + row) * K + kk + c4 * 8;
      const _Float16* s3 = W3t + (size_t)(n0 + row) * K + kk + c4 * 8;
#if ASYNC_LDS
      async_copy16(&B1s[row][c4 * 8], s1);
      async_copy16(&B3s[row][c4 * 8], s3);
#else
      *(uint4*)&B1s[row][c4 * 8] = *(const uint4*)s1;
      *(uint4*)&B3s[row][c4 * 8] = *(const uint4*)s3;
#endif
    }
    wait_async();
    __syncthreads();

#pragma unroll
    for (int kc = 0; kc < 2; ++kc) {
      v16h af = frag_a_lds(&As[0][0], 64, w * 16 + ln, kc * 32, lg);
#pragma unroll
      for (int s = 0; s < 4; ++s) {
        v16h b1 = frag_b_lds(&B1s[s * 16 + ln][kc * 32], lg);
        accU[s] = wmma_f16(af, b1, accU[s]);
        v16h b3 = frag_b_lds(&B3s[s * 16 + ln][kc * 32], lg);
        accT[s] = wmma_f16(af, b3, accT[s]);
      }
    }
    __syncthreads();
  }

#pragma unroll
  for (int s = 0; s < 4; ++s) {
    int n = n0 + s * 16 + ln;
#pragma unroll
    for (int r = 0; r < 8; ++r) {
      int m = m0 + w * 16 + r + 8 * lg;
      float u = accU[s][r], t = accT[s][r];
      float g = (u / (1.f + expf(-u))) * t;
      out[(size_t)m * N + n] = (_Float16)g;
    }
  }
}

// ---------------------------------------------------------------- QKNorm + 2D RoPE
__global__ void __launch_bounds__(256)
qknorm_rope_kernel(_Float16* __restrict__ q, _Float16* __restrict__ k,
                   const float* __restrict__ gq, const float* __restrict__ gk,
                   const int* __restrict__ ridx) {
  int item = blockIdx.x * 8 + (threadIdx.x >> 5);  // (token, head)
  int lane = threadIdx.x & 31;
  int tok = item >> 4;
  int h = item & 15;
  size_t base = (size_t)tok * D_ + (size_t)h * HD_;
  int n = tok % N_;
  int ri = ridx[n];
  bool valid = ri >= 0;
  int safe = valid ? ri : 0;
  float pr = (float)(safe / GRID2), pc = (float)(safe % GRID2);
  int j = lane & 15;
  float invf = powf(10000.f, -(float)j / 16.f);
  float fr = (lane < 16 ? pr : pc) * invf;
  float cs = cosf(fr), sn = sinf(fr);

#pragma unroll
  for (int which = 0; which < 2; ++which) {
    _Float16* p = which ? k : q;
    const float* g = which ? gk : gq;
    float x0 = (float)p[base + lane];
    float x1 = (float)p[base + lane + 32];
    float ss = x0 * x0 + x1 * x1;
#pragma unroll
    for (int off = 16; off; off >>= 1) ss += __shfl_xor(ss, off, 32);
    float rs = rsqrtf(ss / (float)HD_ + EPS_);
    x0 = x0 * rs * g[lane];
    x1 = x1 * rs * g[lane + 32];
    if (valid) {
      float y0 = x0 * cs - x1 * sn;  // rotate_half pairing (d, d+32)
      float y1 = x1 * cs + x0 * sn;
      x0 = y0; x1 = y1;
    }
    p[base + lane] = (_Float16)x0;
    p[base + lane + 32] = (_Float16)x1;
  }
}

// ---------------------------------------------------------------- flash attention
// 4 waves per block; each wave owns a 16-query tile; K/V staged cooperatively.
__global__ void __launch_bounds__(128)
attention_wmma_kernel(const _Float16* __restrict__ q, const _Float16* __restrict__ k,
                      const _Float16* __restrict__ v,
                      const unsigned char* __restrict__ mask,
                      _Float16* __restrict__ out) {
  __shared__ _Float16 Qs[64][64];
  __shared__ _Float16 Ks[32][64];
  __shared__ _Float16 Vst[64][34];     // transposed: [dim][key]
  __shared__ _Float16 Ps[4][16][34];   // per-wave P tile
  const int tid = threadIdx.x;
  const int lane = tid & 31;
  const int wid = tid >> 5;            // 0..3
  const int lg = lane >> 4, ln = lane & 15;
  const int qb = blockIdx.x * 64;      // 64-query block
  const int h = blockIdx.y, bt = blockIdx.z;
  const int q0 = qb + wid * 16;
  const size_t hb = (size_t)bt * N_ * D_ + (size_t)h * HD_;
  const float pscale = 0.125f;  // 1/sqrt(64)

#pragma unroll
  for (int i = 0; i < 4; ++i) {  // stage Q 64x64: 512 x 16B
    int idx = tid + i * 128, row = idx >> 3, c4 = idx & 7;
    const _Float16* src = q + hb + (size_t)(qb + row) * D_ + c4 * 8;
#if ASYNC_LDS
    async_copy16(&Qs[row][c4 * 8], src);
#else
    *(uint4*)&Qs[row][c4 * 8] = *(const uint4*)src;
#endif
  }

  v8f o[4] = {};
  float mrun[8], lrun[8];
#pragma unroll
  for (int r = 0; r < 8; ++r) { mrun[r] = -__builtin_inff(); lrun[r] = 0.f; }

  for (int kb = 0; kb < N_ / 32; ++kb) {
    const int k0 = kb * 32;
#pragma unroll
    for (int i = 0; i < 2; ++i) {  // stage K 32x64: 256 x 16B
      int idx = tid + i * 128, row = idx >> 3, c4 = idx & 7;
      const _Float16* src = k + hb + (size_t)(k0 + row) * D_ + c4 * 8;
#if ASYNC_LDS
      async_copy16(&Ks[row][c4 * 8], src);
#else
      *(uint4*)&Ks[row][c4 * 8] = *(const uint4*)src;
#endif
    }
#pragma unroll 4
    for (int i = 0; i < 16; ++i) {  // stage V transposed
      int idx = tid + i * 128;
      int key = idx >> 6, nd = idx & 63;
      Vst[nd][key] = v[hb + (size_t)(k0 + key) * D_ + nd];
    }
    wait_async();
    __syncthreads();

    // S = Q(16x64) @ K^T(64x32)
    v8f sa[2] = {};
#pragma unroll
    for (int kc = 0; kc < 2; ++kc) {
      v16h af = frag_a_lds(&Qs[0][0], 64, wid * 16 + ln, kc * 32, lg);
#pragma unroll
      for (int s = 0; s < 2; ++s) {
        v16h bf = frag_b_lds(&Ks[s * 16 + ln][kc * 32], lg);
        sa[s] = wmma_f16(af, bf, sa[s]);
      }
    }

    // softcap + mask + online softmax
    float mnew[8];
#pragma unroll
    for (int r = 0; r < 8; ++r) {
      int qg = q0 + r + 8 * lg;
      float rm = mrun[r];
#pragma unroll
      for (int s = 0; s < 2; ++s) {
        float sv = sa[s][r] * pscale;
        sv = CAP_ * tanhf(sv / CAP_);
        int kg = k0 + s * 16 + ln;
        if (!mask[(size_t)qg * N_ + kg]) sv = -__builtin_inff();
        sa[s][r] = sv;
        rm = fmaxf(rm, sv);
      }
#pragma unroll
      for (int off = 1; off < 16; off <<= 1) rm = fmaxf(rm, __shfl_xor(rm, off, 32));
      mnew[r] = rm;
    }

#pragma unroll
    for (int r = 0; r < 8; ++r) {
      bool finite = mnew[r] > -__builtin_inff();
      float alpha = finite ? expf(mrun[r] - mnew[r]) : 0.f;
      float rsum = 0.f;
#pragma unroll
      for (int s = 0; s < 2; ++s) {
        float pv = finite ? expf(sa[s][r] - mnew[r]) : 0.f;
        rsum += pv;
        Ps[wid][r + 8 * lg][s * 16 + ln] = (_Float16)pv;
      }
#pragma unroll
      for (int off = 1; off < 16; off <<= 1) rsum += __shfl_xor(rsum, off, 32);
      lrun[r] = lrun[r] * alpha + rsum;
      mrun[r] = mnew[r];
#pragma unroll
      for (int s = 0; s < 4; ++s) o[s][r] *= alpha;
    }

    // O += P(16x32) @ V(32x64)
    v16h pf = frag_a_lds(&Ps[wid][0][0], 34, ln, 0, lg);
#pragma unroll
    for (int s = 0; s < 4; ++s) {
      v16h bf = frag_b_lds(&Vst[s * 16 + ln][0], lg);
      o[s] = wmma_f16(pf, bf, o[s]);
    }
    __syncthreads();
  }

#pragma unroll
  for (int s = 0; s < 4; ++s)
#pragma unroll
    for (int r = 0; r < 8; ++r) {
      float denom = fmaxf(lrun[r], 1e-20f);
      float val = o[s][r] / denom;
      out[hb + (size_t)(q0 + r + 8 * lg) * D_ + s * 16 + ln] = (_Float16)val;
    }
}

// ---------------------------------------------------------------- launch
extern "C" void kernel_launch(void* const* d_in, const int* in_sizes, int n_in,
                              void* d_out, int out_size, void* d_ws, size_t ws_size,
                              hipStream_t stream) {
  const float* x   = (const float*)d_in[0];
  const float* wq  = (const float*)d_in[1];
  const float* wk  = (const float*)d_in[2];
  const float* wv  = (const float*)d_in[3];
  const float* wo  = (const float*)d_in[4];
  const float* gq  = (const float*)d_in[5];
  const float* gk  = (const float*)d_in[6];
  const float* g1  = (const float*)d_in[7];
  const float* g2  = (const float*)d_in[8];
  const float* w1  = (const float*)d_in[9];
  const float* w3  = (const float*)d_in[10];
  const float* w2  = (const float*)d_in[11];
  const unsigned char* smask = (const unsigned char*)d_in[12];
  const int* ridx  = (const int*)d_in[13];

  char* ws = (char*)d_ws;
  size_t off = 0;
  auto alloc = [&](size_t bytes) {
    void* p = ws + off;
    off += (bytes + 255) & ~(size_t)255;
    return p;
  };
  const size_t SZ_TD_H = (size_t)MTOK_ * D_ * sizeof(_Float16);
  const size_t SZ_TD_F = (size_t)MTOK_ * D_ * sizeof(float);
  const size_t SZ_DD_H = (size_t)D_ * D_ * sizeof(_Float16);
  const size_t SZ_DH_H = (size_t)D_ * HID_ * sizeof(_Float16);
  const size_t SZ_TH_H = (size_t)MTOK_ * HID_ * sizeof(_Float16);

  _Float16* h16   = (_Float16*)alloc(SZ_TD_H);
  _Float16* wqT   = (_Float16*)alloc(SZ_DD_H);   // [N][K] transposed f16 weights
  _Float16* wkT   = (_Float16*)alloc(SZ_DD_H);
  _Float16* wvT   = (_Float16*)alloc(SZ_DD_H);
  _Float16* woT   = (_Float16*)alloc(SZ_DD_H);
  _Float16* w1T   = (_Float16*)alloc(SZ_DH_H);
  _Float16* w3T   = (_Float16*)alloc(SZ_DH_H);
  _Float16* w2T   = (_Float16*)alloc(SZ_DH_H);
  _Float16* q16   = (_Float16*)alloc(SZ_TD_H);
  _Float16* k16   = (_Float16*)alloc(SZ_TD_H);
  _Float16* v16   = (_Float16*)alloc(SZ_TD_H);
  _Float16* ao16  = (_Float16*)alloc(SZ_TD_H);
  float*    xs    = (float*)   alloc(SZ_TD_F);
  _Float16* gbuf  = (_Float16*)alloc(SZ_TH_H);

  // convert + transpose weights: W[K][N] fp32 -> Wt[N][K] f16
  cvt_transpose_kernel<<<dim3(D_/32, D_/32), 256, 0, stream>>>(wq, wqT, D_, D_);
  cvt_transpose_kernel<<<dim3(D_/32, D_/32), 256, 0, stream>>>(wk, wkT, D_, D_);
  cvt_transpose_kernel<<<dim3(D_/32, D_/32), 256, 0, stream>>>(wv, wvT, D_, D_);
  cvt_transpose_kernel<<<dim3(D_/32, D_/32), 256, 0, stream>>>(wo, woT, D_, D_);
  cvt_transpose_kernel<<<dim3(D_/32, HID_/32), 256, 0, stream>>>(w1, w1T, D_, HID_);
  cvt_transpose_kernel<<<dim3(D_/32, HID_/32), 256, 0, stream>>>(w3, w3T, D_, HID_);
  cvt_transpose_kernel<<<dim3(HID_/32, D_/32), 256, 0, stream>>>(w2, w2T, HID_, D_);

  // pre-attention RMSNorm
  rmsnorm_f16_kernel<<<MTOK_, 256, 0, stream>>>(x, g1, h16);

  // QKV projections
  dim3 gproj(D_ / 64, MTOK_ / 128);
  gemm_wmma_kernel<true><<<gproj, 256, 0, stream>>>(h16, wqT, nullptr, q16, MTOK_, D_, D_);
  gemm_wmma_kernel<true><<<gproj, 256, 0, stream>>>(h16, wkT, nullptr, k16, MTOK_, D_, D_);
  gemm_wmma_kernel<true><<<gproj, 256, 0, stream>>>(h16, wvT, nullptr, v16, MTOK_, D_, D_);

  // QK-norm + RoPE
  qknorm_rope_kernel<<<(MTOK_ * NH_) / 8, 256, 0, stream>>>(q16, k16, gq, gk, ridx);

  // attention
  dim3 gatt(N_ / 64, NH_, BT_);
  attention_wmma_kernel<<<gatt, 128, 0, stream>>>(q16, k16, v16, smask, ao16);

  // output projection + residual -> xs (fp32)
  gemm_wmma_kernel<false><<<gproj, 256, 0, stream>>>(ao16, woT, x, xs, MTOK_, D_, D_);

  // FFN: RMSNorm, fused W1/W3 + SiLU gate, W2 + residual -> d_out
  rmsnorm_f16_kernel<<<MTOK_, 256, 0, stream>>>(xs, g2, h16);
  dim3 gffn(HID_ / 64, MTOK_ / 128);
  ffn_gate_wmma_kernel<<<gffn, 256, 0, stream>>>(h16, w1T, w3T, gbuf, MTOK_, HID_, D_);
  gemm_wmma_kernel<false><<<gproj, 256, 0, stream>>>(gbuf, w2T, xs, (float*)d_out,
                                                     MTOK_, D_, HID_);
  (void)in_sizes; (void)n_in; (void)out_size; (void)ws_size;
}
